// TemporalSelfAttention_69758858822510
// MI455X (gfx1250) — compile-verified, matrix operand order checked by reading, not run on hardware
//
#include <hip/hip_runtime.h>
#include <hip/hip_bf16.h>

// ---------------------------------------------------------------------------
// Temporal self-attention with relative positional bias, CDNA5 (gfx1250).
//
// rel term collapses: rel_attn[b,h,i,j] = (sum_t Q[b,h,t,:]).rel_table[clip]/8
// -> precompute r[b,h,k] (32*513) once, gather from LDS in the attention loop.
//
// All matmuls on V_WMMA_F32_16X16X32_BF16 (f32 accumulate).  All operand
// fragments are loaded as 16-byte vectors (the WMMA lane layout is two
// contiguous 8-element runs per lane), V is stored transposed (B,H,DH,T) so
// the P*V B-fragment is contiguous along s.  ~36MB working set is L2-resident
// on MI455X (192MB L2): matrix-unit bound, not HBM bound.
// ---------------------------------------------------------------------------

typedef __attribute__((ext_vector_type(16))) __bf16 v16bf;
typedef __attribute__((ext_vector_type(8)))  __bf16 v8bf;
typedef __attribute__((ext_vector_type(8)))  float  v8f;
typedef __attribute__((ext_vector_type(4)))  float  v4f;

#define BATCH  4
#define SEQ    1024
#define DMODEL 512
#define HEADS  8
#define DHEAD  64
#define NTOK   (BATCH * SEQ)       // 4096
#define MAXREL 256
#define RELN   (2 * MAXREL + 1)    // 513

// Build a 16x32 fragment from a row that is contiguous along k.
// Lane layout (ISA 7.12.2): element e -> k = (e/8)*16 + half*8 + e%8, so the
// fragment is two contiguous 8-element runs at k = half*8 and k = 16+half*8.
__device__ __forceinline__ v16bf frag_from(const __bf16* rowp, int half) {
    const v8bf lo = *(const v8bf*)(rowp + half * 8);        // b128 load
    const v8bf hi = *(const v8bf*)(rowp + 16 + half * 8);   // b128 load
    v16bf f;
#pragma unroll
    for (int e = 0; e < 8; ++e) { f[e] = lo[e]; f[8 + e] = hi[e]; }
    return f;
}

__device__ __forceinline__ v16bf frag_from(const float* rowp, int half) {
    const float* p = rowp + half * 8;
    const v4f a = *(const v4f*)(p);          // 4x b128 loads total
    const v4f b = *(const v4f*)(p + 4);
    const v4f c = *(const v4f*)(p + 16);
    const v4f d = *(const v4f*)(p + 20);
    v16bf f;
#pragma unroll
    for (int e = 0; e < 4; ++e) {
        f[e]      = (__bf16)a[e];
        f[4 + e]  = (__bf16)b[e];
        f[8 + e]  = (__bf16)c[e];
        f[12 + e] = (__bf16)d[e];
    }
    return f;
}

// Output layouts for the projection kernel.
#define LAY_ROW  0   // (NTOK, DMODEL) row-major
#define LAY_BHTD 1   // (B,H,T,DH)  — Q, K
#define LAY_BHDT 2   // (B,H,DH,T)  — V transposed for contiguous P*V loads

// ---------------------------------------------------------------------------
// Y = X * W^T + bias.  One wave per 16x16 output tile, K swept in steps of 32
// (16 WMMAs per tile).
// ---------------------------------------------------------------------------
template <typename TIN, typename TOUT, int LAYOUT>
__global__ __launch_bounds__(256) void proj_kernel(
    const TIN* __restrict__ X, const float* __restrict__ W,
    const float* __restrict__ bias, TOUT* __restrict__ Y) {
    const int wave = threadIdx.x >> 5;
    const int lane = threadIdx.x & 31;
    const int tile = blockIdx.x * 8 + wave;   // 8192 tiles, exact fit
    const int mt   = tile & 31;               // D/16 column tiles
    const int nt   = tile >> 5;               // NTOK/16 row tiles
    const int half = lane >> 4;
    const int nIdx = lane & 15;

    const TIN*   ap = X + (size_t)(nt * 16 + nIdx) * DMODEL;
    const float* bp = W + (size_t)(mt * 16 + nIdx) * DMODEL;

    v8f acc = {};
#pragma unroll 4
    for (int k0 = 0; k0 < DMODEL; k0 += 32) {
        const v16bf a = frag_from(ap + k0, half);
        const v16bf b = frag_from(bp + k0, half);
        acc = __builtin_amdgcn_wmma_f32_16x16x32_bf16(
            false, a, false, b, (short)0, acc, false, false);
    }

    const int   col = mt * 16 + nIdx;
    const float bv  = bias[col];
#pragma unroll
    for (int v = 0; v < 8; ++v) {
        const int   rowg = nt * 16 + half * 8 + v;
        const float val  = acc[v] + bv;
        if (LAYOUT == LAY_ROW) {
            Y[(size_t)rowg * DMODEL + col] = (TOUT)val;
        } else {
            const int b_ = rowg >> 10, t_ = rowg & 1023;
            const int h_ = col >> 6, dh_ = col & 63;
            if (LAYOUT == LAY_BHTD)
                Y[(((size_t)(b_ * HEADS + h_) * SEQ + t_) * DHEAD) + dh_] =
                    (TOUT)val;
            else  // LAY_BHDT
                Y[(((size_t)(b_ * HEADS + h_) * DHEAD + dh_) * SEQ) + t_] =
                    (TOUT)val;
        }
    }
}

// ---------------------------------------------------------------------------
// r[b,h,k] = (sum_t Q[b,h,t,:]) . rel_table[k] / sqrt(DH).  One block per
// (b,h); 513*64 FMAs per block — negligible.
// ---------------------------------------------------------------------------
__global__ __launch_bounds__(256) void rel_kernel(
    const __bf16* __restrict__ Q, const float* __restrict__ relTab,
    float* __restrict__ R) {
    const int bh  = blockIdx.x;   // 32
    const int tid = threadIdx.x;
    __shared__ float ps[4][DHEAD];
    __shared__ float qsum[DHEAD];

    const int d = tid & 63, c = tid >> 6;
    float s = 0.f;
    const __bf16* qp = Q + (size_t)bh * SEQ * DHEAD + d;
    for (int t = c * 256; t < c * 256 + 256; ++t)
        s += (float)qp[(size_t)t * DHEAD];
    ps[c][d] = s;
    __syncthreads();
    if (tid < DHEAD)
        qsum[tid] = ps[0][tid] + ps[1][tid] + ps[2][tid] + ps[3][tid];
    __syncthreads();

    const float inv_scale = 0.125f;   // 1/sqrt(64)
    for (int k = tid; k < RELN; k += 256) {
        float dot = 0.f;
#pragma unroll
        for (int dd = 0; dd < DHEAD; ++dd)
            dot += qsum[dd] * relTab[k * DHEAD + dd];
        R[bh * RELN + k] = dot * inv_scale;
    }
}

// ---------------------------------------------------------------------------
// Flash attention.  One wave per 16-query tile; 32 keys per step:
// 4 WMMAs for S (two 16x16 tiles, K=64) + 4 WMMAs for P*V (one per DH tile).
// All 8 waves of a block share one (b,h), so the 513-entry rel row is staged
// in LDS once per block.  P is re-laid-out D->A through a 1KB/wave LDS slab
// behind an explicit gfx1250 s_wait_dscnt fence.
// ---------------------------------------------------------------------------
__global__ __launch_bounds__(256) void attn_kernel(
    const __bf16* __restrict__ Q, const __bf16* __restrict__ K,
    const __bf16* __restrict__ Vt, const float* __restrict__ R,
    const unsigned char* __restrict__ mask, __bf16* __restrict__ ctx) {
    __shared__ __bf16 pbuf[8][16][32];   // per-wave P staging (8KB)
    __shared__ float  rbuf[RELN];        // rel bias row for this (b,h)

    const int wave = threadIdx.x >> 5;
    const int lane = threadIdx.x & 31;
    const int bh   = blockIdx.x >> 3;                  // 8 blocks per (b,h)
    const int qt   = (blockIdx.x & 7) * 8 + wave;      // 64 q-tiles per (b,h)
    const int b    = bh >> 3;
    const int h    = bh & 7;
    const int qbase = qt * 16;
    const int half  = lane >> 4;
    const int nIdx  = lane & 15;

    const __bf16* Qb  = Q  + (size_t)bh * SEQ * DHEAD;
    const __bf16* Kb  = K  + (size_t)bh * SEQ * DHEAD;
    const __bf16* Vtb = Vt + (size_t)bh * DHEAD * SEQ;
    const unsigned char* mb = mask + b * SEQ;

    // stage rel row into LDS (shared by the whole block)
    for (int i = threadIdx.x; i < RELN; i += 256)
        rbuf[i] = R[bh * RELN + i];
    __syncthreads();

    // Q A-fragments for d=0..31 and 32..63 (4x b128 loads, held in regs)
    const __bf16* qp = Qb + (size_t)(qbase + nIdx) * DHEAD;
    const v16bf qa0 = frag_from(qp, half);
    const v16bf qa1 = frag_from(qp + 32, half);

    v8f o0 = {}, o1 = {}, o2 = {}, o3 = {};
    float rowM[8], rowL[8];
#pragma unroll
    for (int v = 0; v < 8; ++v) { rowM[v] = -3.0e38f; rowL[v] = 0.f; }

    const float inv_scale = 0.125f;

    for (int s0 = 0; s0 < SEQ; s0 += 32) {
        // ---- S = Q K^T for two 16-key tiles -------------------------------
        v8f s_t[2];
#pragma unroll
        for (int t = 0; t < 2; ++t) {
            const __bf16* kp = Kb + (size_t)(s0 + t * 16 + nIdx) * DHEAD;
            const v16bf kb0 = frag_from(kp, half);
            const v16bf kb1 = frag_from(kp + 32, half);
            v8f acc = {};
            acc = __builtin_amdgcn_wmma_f32_16x16x32_bf16(
                false, qa0, false, kb0, (short)0, acc, false, false);
            acc = __builtin_amdgcn_wmma_f32_16x16x32_bf16(
                false, qa1, false, kb1, (short)0, acc, false, false);
            s_t[t] = acc;
        }

        // ---- logits: scale + relative bias (LDS gather) + key mask --------
        const int  j0 = s0 + nIdx;
        const int  j1 = s0 + 16 + nIdx;
        const bool m0 = mb[j0] != 0;
        const bool m1 = mb[j1] != 0;
        float lg0[8], lg1[8];
#pragma unroll
        for (int v = 0; v < 8; ++v) {
            const int iRow = qbase + half * 8 + v;
            int d0 = j0 - iRow; d0 = min(max(d0, -MAXREL), MAXREL) + MAXREL;
            int d1 = j1 - iRow; d1 = min(max(d1, -MAXREL), MAXREL) + MAXREL;
            const float l0 = s_t[0][v] * inv_scale + rbuf[d0];
            const float l1 = s_t[1][v] * inv_scale + rbuf[d1];
            lg0[v] = m0 ? l0 : -1.0e30f;
            lg1[v] = m1 ? l1 : -1.0e30f;
        }

        // ---- online softmax update ---------------------------------------
        float p0[8], p1[8], alpha[8];
#pragma unroll
        for (int v = 0; v < 8; ++v) {
            float mx = fmaxf(lg0[v], lg1[v]);
#pragma unroll
            for (int msk = 1; msk <= 8; msk <<= 1)
                mx = fmaxf(mx, __shfl_xor(mx, msk, 32));
            const float newM = fmaxf(rowM[v], mx);
            alpha[v] = __expf(rowM[v] - newM);
            p0[v] = __expf(lg0[v] - newM);
            p1[v] = __expf(lg1[v] - newM);
            float rs = p0[v] + p1[v];
#pragma unroll
            for (int msk = 1; msk <= 8; msk <<= 1)
                rs += __shfl_xor(rs, msk, 32);
            rowL[v] = rowL[v] * alpha[v] + rs;
            rowM[v] = newM;
        }
#pragma unroll
        for (int v = 0; v < 8; ++v) {
            o0[v] *= alpha[v]; o1[v] *= alpha[v];
            o2[v] *= alpha[v]; o3[v] *= alpha[v];
        }

        // ---- stage P (D-layout -> A-layout) through LDS -------------------
#pragma unroll
        for (int v = 0; v < 8; ++v) {
            pbuf[wave][half * 8 + v][nIdx]      = (__bf16)p0[v];
            pbuf[wave][half * 8 + v][16 + nIdx] = (__bf16)p1[v];
        }
        asm volatile("s_wait_dscnt 0" ::: "memory");   // cross-lane LDS RAW
        const v16bf pa = frag_from(&pbuf[wave][nIdx][0], half);

        // ---- O += P * V   (V transposed: contiguous b128 loads) -----------
#pragma unroll
        for (int dt = 0; dt < 4; ++dt) {
            const __bf16* vp =
                Vtb + (size_t)(dt * 16 + nIdx) * SEQ + s0;
            const v16bf vb = frag_from(vp, half);
            v8f& o = (dt == 0 ? o0 : dt == 1 ? o1 : dt == 2 ? o2 : o3);
            o = __builtin_amdgcn_wmma_f32_16x16x32_bf16(
                false, pa, false, vb, (short)0, o, false, false);
        }
    }

    // ---- normalize and write ctx in (B,T,D) bf16 layout -------------------
#pragma unroll
    for (int v = 0; v < 8; ++v) {
        const float invL = 1.0f / rowL[v];
        const int   tok  = qbase + half * 8 + v;
        __bf16* cp = ctx + ((size_t)b * SEQ + tok) * DMODEL + h * DHEAD + nIdx;
        cp[0]  = (__bf16)(o0[v] * invL);
        cp[16] = (__bf16)(o1[v] * invL);
        cp[32] = (__bf16)(o2[v] * invL);
        cp[48] = (__bf16)(o3[v] * invL);
    }
}

// ---------------------------------------------------------------------------
extern "C" void kernel_launch(void* const* d_in, const int* in_sizes, int n_in,
                              void* d_out, int out_size, void* d_ws,
                              size_t ws_size, hipStream_t stream) {
    (void)in_sizes; (void)n_in; (void)out_size; (void)ws_size;

    const float*         x    = (const float*)d_in[0];
    const unsigned char* mask = (const unsigned char*)d_in[1];
    const float*         Wq   = (const float*)d_in[2];
    const float*         bq   = (const float*)d_in[3];
    const float*         Wk   = (const float*)d_in[4];
    const float*         bk   = (const float*)d_in[5];
    const float*         Wv   = (const float*)d_in[6];
    const float*         bv   = (const float*)d_in[7];
    const float*         Wo   = (const float*)d_in[8];
    const float*         bo   = (const float*)d_in[9];
    const float*         relT = (const float*)d_in[10];

    char* ws = (char*)d_ws;
    const size_t QKV_BYTES = (size_t)NTOK * DMODEL * sizeof(__bf16);  // 4MB
    __bf16* q   = (__bf16*)(ws + 0 * QKV_BYTES);
    __bf16* k   = (__bf16*)(ws + 1 * QKV_BYTES);
    __bf16* vt  = (__bf16*)(ws + 2 * QKV_BYTES);   // (B,H,DH,T)
    __bf16* ctx = (__bf16*)(ws + 3 * QKV_BYTES);
    float*  r   = (float*)(ws + 4 * QKV_BYTES);    // 32*513 floats

    const dim3 pg(1024), pb(256);
    proj_kernel<float, __bf16, LAY_BHTD><<<pg, pb, 0, stream>>>(x, Wq, bq, q);
    proj_kernel<float, __bf16, LAY_BHTD><<<pg, pb, 0, stream>>>(x, Wk, bk, k);
    proj_kernel<float, __bf16, LAY_BHDT><<<pg, pb, 0, stream>>>(x, Wv, bv, vt);
    rel_kernel<<<dim3(32), pb, 0, stream>>>(q, relT, r);
    attn_kernel<<<dim3(256), pb, 0, stream>>>(q, k, vt, r, mask, ctx);
    proj_kernel<__bf16, float, LAY_ROW><<<pg, pb, 0, stream>>>(ctx, Wo, bo,
                                                               (float*)d_out);
}
